// PairwiseIsActionTakenBLSTM_8555574853824
// MI455X (gfx1250) — matrix operand order repaired and checked by reading, not compiled, not run on hardware
//
#include <hip/hip_runtime.h>
#include <math.h>

typedef __attribute__((ext_vector_type(16))) _Float16 v16h;
typedef __attribute__((ext_vector_type(8)))  _Float16 v8h;
typedef __attribute__((ext_vector_type(8)))  float    v8f;
typedef __attribute__((ext_vector_type(4)))  float    v4f;

// ---------------- packed-weight tile bases (in 512-half tiles) ----------------
#define TB_FC1   0      // N=128 K=256  -> 8x8 = 64 tiles
#define TB_FC1A  64     // N=128 K=164->192 -> 8x6 = 48
#define TB_FC1B  112    // N=128 K=128 -> 8x4 = 32
#define TB_FC1C  144    // N=64  K=140->160 -> 4x5 = 20
#define TB_FC2   164    // N=32  K=64  -> 2x2 = 4
#define TB_FC3   168    // N=32  K=32  -> 2x1 = 2
#define TB_WIH   170    // N=128 K=32  -> 8x1 = 8
#define TB_WHH   178    // N=128 K=32  -> 8x1 = 8
#define TB_FC4   186    // N=32  K=32  -> 2x1 = 2
#define TOTAL_TILES 188

struct PackDesc { const float* w; int Kreal; int ntiles; int kchunks; int tile_base; };
struct PackArgs { PackDesc d[9]; };

// Repack f32 weights [N,K] row-major into f16 WMMA A-fragment order.
// Tile = (kchunk*ntiles + ntile), 512 halfs per tile laid out lane-major:
// lane (hi,r), half i -> A element (m = 16*ntile + r,
//   k = 32*kchunk + 16*hi + i)   [contiguous k order, matches B fragments]
__global__ void pack_weights_kernel(PackArgs pa, _Float16* __restrict__ outp) {
  int gid  = blockIdx.x * blockDim.x + threadIdx.x;
  int tile = gid >> 9;
  if (tile >= TOTAL_TILES) return;
  int e = gid & 511;
  int L = 0;
#pragma unroll
  for (int j = 1; j < 9; ++j) if (tile >= pa.d[j].tile_base) L = j;
  PackDesc D = pa.d[L];
  int lt     = tile - D.tile_base;
  int kchunk = lt / D.ntiles;
  int ntile  = lt - kchunk * D.ntiles;
  int lane = e >> 4, i = e & 15;
  int hi = lane >> 4, r = lane & 15;
  int klocal = 16 * hi + i;
  int m = 16 * ntile + r;
  int k = 32 * kchunk + klocal;
  float v = (k < D.Kreal) ? D.w[(size_t)m * D.Kreal + k] : 0.0f;
  outp[(size_t)tile * 512 + e] = (_Float16)v;
}

// ---------------- main fused kernel helpers ----------------
__device__ __forceinline__ v8f wmma16(v16h a, v16h b, v8f c) {
  return __builtin_amdgcn_wmma_f32_16x16x32_f16(false, a, false, b, (short)0, c,
                                                false, false);
}

// 16 contiguous f32 -> 16 f16 (B fragment half for this lane)
__device__ __forceinline__ v16h cvt16(const float* p) {
  v4f f0 = ((const v4f*)p)[0];
  v4f f1 = ((const v4f*)p)[1];
  v4f f2 = ((const v4f*)p)[2];
  v4f f3 = ((const v4f*)p)[3];
  v16h r;
#pragma unroll
  for (int i = 0; i < 4; ++i) {
    r[i]      = (_Float16)f0[i];
    r[4 + i]  = (_Float16)f1[i];
    r[8 + i]  = (_Float16)f2[i];
    r[12 + i] = (_Float16)f3[i];
  }
  return r;
}

__device__ __forceinline__ float sigf(float x) { return 1.0f / (1.0f + __expf(-x)); }

template <int NT>
__device__ __forceinline__ void init_acc(v8f* acc, const float* bias, int mbase) {
#pragma unroll
  for (int t = 0; t < NT; ++t)
#pragma unroll
    for (int v = 0; v < 8; ++v) acc[t][v] = bias[16 * t + mbase + v];
}

// GEMM with B fragments read from this wave's LDS activation row.
template <int NT, int KC>
__device__ __forceinline__ void layer_lds(v8f* acc, const _Float16* lw,
                                          const _Float16* brow, int lane) {
#pragma unroll
  for (int kk = 0; kk < KC; ++kk) {
    v16h b = *(const v16h*)(brow + kk * 32);
#pragma unroll
    for (int t = 0; t < NT; ++t) {
      v16h a = *(const v16h*)(lw + (size_t)(kk * NT + t) * 512 + lane * 16);
      acc[t] = wmma16(a, b, acc[t]);
    }
  }
}

// ReLU + f32->f16 + 16B LDS store per tile (feats 16t+8hi .. +8 of row n)
template <int NT>
__device__ __forceinline__ void store_relu(const v8f* acc, _Float16* drow) {
#pragma unroll
  for (int t = 0; t < NT; ++t) {
    v8h hv;
#pragma unroll
    for (int v = 0; v < 8; ++v) hv[v] = (_Float16)fmaxf(acc[t][v], 0.0f);
    *(v8h*)(drow + 16 * t) = hv;
  }
}

#define ACT_STRIDE 192  // padded feature count (164 -> 192), keeps 32B alignment

__global__ __launch_bounds__(256) void fused_mlp_lstm(
    const float* __restrict__ x, const float* __restrict__ w,
    const float* __restrict__ h0, const float* __restrict__ c0,
    const _Float16* __restrict__ pk,
    const float* __restrict__ fc1_b, const float* __restrict__ fc1a_b,
    const float* __restrict__ fc1b_b, const float* __restrict__ fc1c_b,
    const float* __restrict__ fc2_b, const float* __restrict__ fc3_b,
    const float* __restrict__ b_ih, const float* __restrict__ b_hh,
    const float* __restrict__ fc4_b, const float* __restrict__ fc5_b,
    const float* __restrict__ fc5_w, const float* __restrict__ w_game,
    float* __restrict__ out) {
  __shared__ __align__(64) _Float16 act[8][16][ACT_STRIDE];  // 48 KB, per-wave rows
  __shared__ float cp[848];                                  // biases + fc5_w + w_game

  const int tid = threadIdx.x;
  // cooperative preload of small params into LDS
  for (int i = tid; i < 845; i += 256) {
    float v;
    if      (i < 128) v = fc1_b[i];
    else if (i < 256) v = fc1a_b[i - 128];
    else if (i < 384) v = fc1b_b[i - 256];
    else if (i < 448) v = fc1c_b[i - 384];
    else if (i < 480) v = fc2_b[i - 448];
    else if (i < 512) v = fc3_b[i - 480];
    else if (i < 640) v = b_ih[i - 512];
    else if (i < 768) v = b_hh[i - 640];
    else if (i < 800) v = fc4_b[i - 768];
    else if (i == 800) v = fc5_b[0];
    else if (i < 833) v = fc5_w[i - 801];
    else              v = w_game[i - 833];
    cp[i] = v;
  }
  __syncthreads();

  const int wid = tid >> 5, lane = tid & 31;
  const int n = lane & 15, hi = lane >> 4;
  const long base = (long)blockIdx.x * 128 + wid * 16;
  const long row  = base + n;

  _Float16* myrow = &act[wid][n][0];
  const _Float16* brow = myrow + 16 * hi;  // B-fragment read base (lane n,hi)
  _Float16* drow       = myrow + 8 * hi;   // D-fragment store base

  v8f acc[8];

  // ---- fc1: [B,256] -> 128, B frags straight from global x ----
  init_acc<8>(acc, &cp[0], 8 * hi);
  {
    const float* xr = x + row * 256 + hi * 16;
    const _Float16* lw = pk + (size_t)TB_FC1 * 512;
#pragma unroll
    for (int kk = 0; kk < 8; ++kk) {
      v16h b = cvt16(xr + kk * 32);
#pragma unroll
      for (int t = 0; t < 8; ++t) {
        v16h a = *(const v16h*)(lw + (size_t)(kk * 8 + t) * 512 + lane * 16);
        acc[t] = wmma16(a, b, acc[t]);
      }
    }
  }
  store_relu<8>(acc, drow);
  // concat w -> feats 128..163, zero-pad 164..191
  {
    const float* wr = w + row * 36;
    for (int j = 18 * hi; j < 18 * hi + 18; ++j) myrow[128 + j] = (_Float16)wr[j];
    for (int j = 164 + 14 * hi; j < 164 + 14 * hi + 14; ++j) myrow[j] = (_Float16)0.0f;
  }

  // ---- fc1a: 164(->192) -> 128 ----
  init_acc<8>(acc, &cp[128], 8 * hi);
  layer_lds<8, 6>(acc, pk + (size_t)TB_FC1A * 512, brow, lane);
  store_relu<8>(acc, drow);

  // ---- fc1b: 128 -> 128 ----
  init_acc<8>(acc, &cp[256], 8 * hi);
  layer_lds<8, 4>(acc, pk + (size_t)TB_FC1B * 512, brow, lane);
  store_relu<8>(acc, drow);
  // concat broadcast w_game -> feats 128..139, zero-pad 140..159
  {
    for (int j = 6 * hi; j < 6 * hi + 6; ++j) myrow[128 + j] = (_Float16)cp[833 + j];
    for (int j = 140 + 10 * hi; j < 140 + 10 * hi + 10; ++j) myrow[j] = (_Float16)0.0f;
  }

  // ---- fc1c: 140(->160) -> 64 ----
  init_acc<4>(acc, &cp[384], 8 * hi);
  layer_lds<4, 5>(acc, pk + (size_t)TB_FC1C * 512, brow, lane);
  store_relu<4>(acc, drow);

  // ---- fc2: 64 -> 32 ----
  init_acc<2>(acc, &cp[448], 8 * hi);
  layer_lds<2, 2>(acc, pk + (size_t)TB_FC2 * 512, brow, lane);
  store_relu<2>(acc, drow);

  // ---- fc3: 32 -> 32 ----
  init_acc<2>(acc, &cp[480], 8 * hi);
  layer_lds<2, 1>(acc, pk + (size_t)TB_FC3 * 512, brow, lane);
  store_relu<2>(acc, drow);

  // ---- LSTM gates: h@w_ih^T + h0@w_hh^T + b_ih + b_hh, N=128 ----
  {
    const int mb = 8 * hi;
#pragma unroll
    for (int t = 0; t < 8; ++t)
#pragma unroll
      for (int v = 0; v < 8; ++v)
        acc[t][v] = cp[512 + 16 * t + mb + v] + cp[640 + 16 * t + mb + v];

    v16h bh = *(const v16h*)brow;  // fc3 output from LDS
    const _Float16* lwih = pk + (size_t)TB_WIH * 512;
#pragma unroll
    for (int t = 0; t < 8; ++t) {
      v16h a = *(const v16h*)(lwih + (size_t)t * 512 + lane * 16);
      acc[t] = wmma16(a, bh, acc[t]);
    }
    v16h bh0 = cvt16(h0 + row * 32 + hi * 16);  // h0 straight from global
    const _Float16* lwhh = pk + (size_t)TB_WHH * 512;
#pragma unroll
    for (int t = 0; t < 8; ++t) {
      v16h a = *(const v16h*)(lwhh + (size_t)t * 512 + lane * 16);
      acc[t] = wmma16(a, bh0, acc[t]);
    }
  }

  // ---- LSTM elementwise: gates tiles {i:0-1, f:2-3, g:4-5, o:6-7} ----
  {
    const float* c0r = c0 + row * 32;
#pragma unroll
    for (int t2 = 0; t2 < 2; ++t2) {
      v4f ca = ((const v4f*)(c0r + 16 * t2 + 8 * hi))[0];
      v4f cb = ((const v4f*)(c0r + 16 * t2 + 8 * hi))[1];
      v8h hv;
#pragma unroll
      for (int v = 0; v < 8; ++v) {
        float cv = (v < 4) ? ca[v] : cb[v - 4];
        float ig = acc[t2][v], fg = acc[2 + t2][v];
        float gg = acc[4 + t2][v], og = acc[6 + t2][v];
        float cn = sigf(fg) * cv + sigf(ig) * tanhf(gg);
        hv[v] = (_Float16)(sigf(og) * tanhf(cn));
      }
      *(v8h*)(drow + 16 * t2) = hv;  // h_new -> LDS feats 0..31
    }
  }

  // ---- fc4: 32 -> 32 ----
  init_acc<2>(acc, &cp[768], 8 * hi);
  layer_lds<2, 1>(acc, pk + (size_t)TB_FC4 * 512, brow, lane);
  store_relu<2>(acc, drow);

  // ---- fc5 + sigmoid: lanes 0..15 reduce their row ----
  if (hi == 0) {
    float s = cp[800];
#pragma unroll
    for (int j = 0; j < 32; ++j) s += (float)myrow[j] * cp[801 + j];
    out[row] = sigf(s);
  }
}

extern "C" void kernel_launch(void* const* d_in, const int* in_sizes, int n_in,
                              void* d_out, int out_size, void* d_ws, size_t ws_size,
                              hipStream_t stream) {
  const float* x      = (const float*)d_in[0];
  const float* w      = (const float*)d_in[1];
  const float* h0     = (const float*)d_in[2];
  const float* c0     = (const float*)d_in[3];
  const float* fc1_w  = (const float*)d_in[4];
  const float* fc1_b  = (const float*)d_in[5];
  const float* fc1a_w = (const float*)d_in[6];
  const float* fc1a_b = (const float*)d_in[7];
  const float* fc1b_w = (const float*)d_in[8];
  const float* fc1b_b = (const float*)d_in[9];
  const float* w_game = (const float*)d_in[10];
  const float* fc1c_w = (const float*)d_in[11];
  const float* fc1c_b = (const float*)d_in[12];
  const float* fc2_w  = (const float*)d_in[13];
  const float* fc2_b  = (const float*)d_in[14];
  const float* fc3_w  = (const float*)d_in[15];
  const float* fc3_b  = (const float*)d_in[16];
  const float* w_ih   = (const float*)d_in[17];
  const float* w_hh   = (const float*)d_in[18];
  const float* b_ih   = (const float*)d_in[19];
  const float* b_hh   = (const float*)d_in[20];
  const float* fc4_w  = (const float*)d_in[21];
  const float* fc4_b  = (const float*)d_in[22];
  const float* fc5_w  = (const float*)d_in[23];
  const float* fc5_b  = (const float*)d_in[24];
  float* out    = (float*)d_out;
  _Float16* pk  = (_Float16*)d_ws;  // ~193 KB of packed f16 weights

  PackArgs pa;
  pa.d[0] = {fc1_w, 256, 8, 8, TB_FC1};
  pa.d[1] = {fc1a_w, 164, 8, 6, TB_FC1A};
  pa.d[2] = {fc1b_w, 128, 8, 4, TB_FC1B};
  pa.d[3] = {fc1c_w, 140, 4, 5, TB_FC1C};
  pa.d[4] = {fc2_w, 64, 2, 2, TB_FC2};
  pa.d[5] = {fc3_w, 32, 2, 1, TB_FC3};
  pa.d[6] = {w_ih, 32, 8, 1, TB_WIH};
  pa.d[7] = {w_hh, 32, 8, 1, TB_WHH};
  pa.d[8] = {fc4_w, 32, 2, 1, TB_FC4};

  const int total_halfs = TOTAL_TILES * 512;
  pack_weights_kernel<<<(total_halfs + 255) / 256, 256, 0, stream>>>(pa, pk);

  const int B = in_sizes[0] / 256;  // 131072
  fused_mlp_lstm<<<B / 128, 256, 0, stream>>>(
      x, w, h0, c0, pk, fc1_b, fc1a_b, fc1b_b, fc1c_b, fc2_b, fc3_b, b_ih, b_hh,
      fc4_b, fc5_b, fc5_w, w_game, out);
}
